// GCN_3092376453711
// MI455X (gfx1250) — matrix operand order repaired and checked by reading, not compiled, not run on hardware
//
#include <hip/hip_runtime.h>
#include <hip/hip_bf16.h>

// Problem constants (match reference)
constexpr int Nn  = 50000;
constexpr int Ee  = 800000;
constexpr int FIN = 1280;
constexpr int Hh  = 256;
constexpr int Gg  = 64;
constexpr int Cc  = 2752;
constexpr float EPSb = 1e-5f;

typedef __attribute__((ext_vector_type(16))) _Float16 v16h;
typedef __attribute__((ext_vector_type(8)))  _Float16 v8h;
typedef __attribute__((ext_vector_type(8)))  float    v8f;

// ---------------------------------------------------------------------------
// Utility kernels
// ---------------------------------------------------------------------------
__global__ void zero_kernel(float* p, int n) {
    int i = blockIdx.x * 256 + threadIdx.x;
    if (i < n) p[i] = 0.0f;
}

// Transposed f32 -> f16 weight convert: whT[n][k] = w[k][n]  (whT is [Hh][K])
__global__ void cvt_f16_T_kernel(const float* __restrict__ w, _Float16* __restrict__ whT, int K) {
    int idx = blockIdx.x * 256 + threadIdx.x;   // over Hh*K, output-linear (coalesced stores)
    if (idx >= Hh * K) return;
    int n = idx / K;
    int k = idx - n * K;
    whT[idx] = (_Float16)w[(size_t)k * Hh + n];
}

__global__ void deg_kernel(const int* __restrict__ dst, float* __restrict__ deg) {
    int e = blockIdx.x * 256 + threadIdx.x;
    if (e < Ee) atomicAdd(&deg[dst[e]], 1.0f);
}

__global__ void dinv_kernel(float* deg) {
    int i = blockIdx.x * 256 + threadIdx.x;
    if (i < Nn) deg[i] = rsqrtf(deg[i] + 1.0f);   // deg^{-1/2} with self-loop
}

// ---------------------------------------------------------------------------
// Tiled WMMA GEMM:  C[M,Ndim] = A[M,K](f32) * B(f16, pre-transposed [Ndim][K])
// Block: 256 threads (8 waves), tile 128x128, BK=32.
// Wave grid 4x2; each wave owns a 32x64 sub-tile = 2x4 WMMA 16x16 tiles.
// A converted f32->f16 into LDS on the fly (clamped addresses, no exec
// branches so the 4 tile loads pipeline); B tile staged with
// global_load_async_to_lds_b128 (ASYNCcnt), drained by s_wait_asynccnt.
// ---------------------------------------------------------------------------
constexpr int BM = 128, BN = 128, BK = 32, APAD = 8;

__global__ __launch_bounds__(256)
void gemm_wmma_f16(const float* __restrict__ A, const _Float16* __restrict__ BhT,
                   float* __restrict__ Cout, int Mdim, int Kdim, int Ndim) {
    __shared__ _Float16 As[BM][BK + APAD];   // row-major A tile (f16)
    __shared__ _Float16 Bs[BN][BK + APAD];   // Bs[n][k] (already K-major in global)

    const int tid  = threadIdx.x;
    const int lane = tid & 31;
    const int wid  = tid >> 5;        // 0..7 (wave32)
    const int wr   = wid >> 1;        // 0..3
    const int wc   = wid & 1;         // 0..1
    const int m0   = blockIdx.x * BM;
    const int n0   = blockIdx.y * BN;

    v8f acc[2][4];
    const v8f vzero = {};
#pragma unroll
    for (int i = 0; i < 2; ++i)
#pragma unroll
        for (int j = 0; j < 4; ++j) acc[i][j] = vzero;

    const int aCol = (tid & 7) * 4;   // 0,4,...,28
    const int aRow = tid >> 3;        // 0..31
    const int half = lane >> 4;       // lane-half select
    const int lm   = lane & 15;

    for (int k0 = 0; k0 < Kdim; k0 += BK) {
        // ---- B tile via async global->LDS copy: 128 rows x 64B = 512 x 16B ----
#pragma unroll
        for (int c = tid; c < BN * 4; c += 256) {
            int n    = c >> 2;
            int part = c & 3;
            const _Float16* gp = &BhT[(size_t)(n0 + n) * Kdim + k0 + part * 8];
            unsigned lds = (unsigned)(unsigned long long)(size_t)&Bs[n][part * 8];
            asm volatile("global_load_async_to_lds_b128 %0, %1, off"
                         :: "v"(lds), "v"((unsigned long long)(size_t)gp)
                         : "memory");
        }

        // ---- A tile: 128x32 f32 -> f16 in LDS; clamped rows, batched loads ----
        float4 vv[4];
#pragma unroll
        for (int r = 0; r < 4; ++r) {
            int gm = m0 + aRow + r * 32;
            if (gm > Mdim - 1) gm = Mdim - 1;    // clamp: junk rows never stored
            vv[r] = *(const float4*)&A[(size_t)gm * Kdim + k0 + aCol];
        }
#pragma unroll
        for (int r = 0; r < 4; ++r) {
            int m = aRow + r * 32;
            As[m][aCol + 0] = (_Float16)vv[r].x;
            As[m][aCol + 1] = (_Float16)vv[r].y;
            As[m][aCol + 2] = (_Float16)vv[r].z;
            As[m][aCol + 3] = (_Float16)vv[r].w;
        }

        // drain async B copies, then block barrier
        asm volatile("s_wait_asynccnt 0" ::: "memory");
        __syncthreads();

        // Prefetch next A tile (global_prefetch_b8)
        if (k0 + BK < Kdim) {
            int gm = m0 + aRow;
            if (gm < Mdim) __builtin_prefetch(&A[(size_t)gm * Kdim + k0 + BK + aCol], 0, 0);
        }

        // ---- fragments ----
        v16h afrag[2], bfrag[4];
        const int abase = half * 8;            // A: K offset 0/8 within each 16-chunk
        const int bbase = half * 8;            // B: K offset 0/8 within each 16-chunk
#pragma unroll
        for (int mi = 0; mi < 2; ++mi) {
            int m = wr * 32 + mi * 16 + lm;
            v8h lo = *(const v8h*)&As[m][abase];
            v8h hi = *(const v8h*)&As[m][16 + abase];
            v16h f;
#pragma unroll
            for (int i = 0; i < 8; ++i) { f[i] = lo[i]; f[i + 8] = hi[i]; }
            afrag[mi] = f;
        }
#pragma unroll
        for (int ni = 0; ni < 4; ++ni) {
            int n = wc * 64 + ni * 16 + lm;
            v8h lo = *(const v8h*)&Bs[n][bbase];
            v8h hi = *(const v8h*)&Bs[n][16 + bbase];
            v16h f;
#pragma unroll
            for (int i = 0; i < 8; ++i) { f[i] = lo[i]; f[i + 8] = hi[i]; }
            bfrag[ni] = f;
        }
        // ---- 8 WMMA per wave per k-step ----
#pragma unroll
        for (int mi = 0; mi < 2; ++mi)
#pragma unroll
            for (int ni = 0; ni < 4; ++ni)
                acc[mi][ni] = __builtin_amdgcn_wmma_f32_16x16x32_f16(
                    false, afrag[mi], false, bfrag[ni],
                    (short)0, acc[mi][ni], false, false);
        __syncthreads();
    }

    // ---- store: VGPR v -> M = v + 8*lanehalf within 16x16 tile ----
#pragma unroll
    for (int mi = 0; mi < 2; ++mi)
#pragma unroll
        for (int ni = 0; ni < 4; ++ni)
#pragma unroll
            for (int v = 0; v < 8; ++v) {
                int row = m0 + wr * 32 + mi * 16 + v + half * 8;
                int col = n0 + wc * 64 + ni * 16 + lm;
                if (row < Mdim) Cout[(size_t)row * Ndim + col] = acc[mi][ni][v];
            }
}

// ---------------------------------------------------------------------------
// GCN aggregation
// ---------------------------------------------------------------------------
// agg[i,c] = h[i,c] * dinv[i]^2 + bias[c]   (self-loop + bias)
__global__ void init_agg_kernel(const float* __restrict__ h, const float* __restrict__ dinv,
                                const float* __restrict__ b, float* __restrict__ agg) {
    int i = blockIdx.x;       // node
    int c = threadIdx.x;      // channel (H = 256 = blockDim)
    float di = dinv[i];
    agg[(size_t)i * Hh + c] = h[(size_t)i * Hh + c] * (di * di) + b[c];
}

// edge scatter: agg[dst] += h[src] * dinv[src]*dinv[dst]; one thread = 4 channels
__global__ void scatter_kernel(const int* __restrict__ src, const int* __restrict__ dst,
                               const float* __restrict__ dinv, const float* __restrict__ h,
                               float* __restrict__ agg) {
    int t = blockIdx.x * 256 + threadIdx.x;   // E * 64 threads
    if (t >= Ee * (Hh / 4)) return;
    int e = t >> 6;                 // /64
    int c = (t & 63) * 4;
    int s = src[e], d = dst[e];
    float w = dinv[s] * dinv[d];
    float4 v = *(const float4*)&h[(size_t)s * Hh + c];
    float* p = &agg[(size_t)d * Hh + c];
    atomicAdd(p + 0, v.x * w);
    atomicAdd(p + 1, v.y * w);
    atomicAdd(p + 2, v.z * w);
    atomicAdd(p + 3, v.w * w);
}

// ---------------------------------------------------------------------------
// BatchNorm (ReLU fused via flag; ReLU recomputed in apply to skip a pass)
// ---------------------------------------------------------------------------
constexpr int BN_ROWS = 128;
__global__ void bn_stats_kernel(const float* __restrict__ a, float* __restrict__ stats, int doRelu) {
    int c = threadIdx.x;                      // channel, blockDim = H
    int r0 = blockIdx.x * BN_ROWS;
    float s = 0.f, s2 = 0.f;
    for (int r = 0; r < BN_ROWS; ++r) {
        int i = r0 + r;
        if (i >= Nn) break;
        float v = a[(size_t)i * Hh + c];
        if (doRelu) v = fmaxf(v, 0.f);
        s += v; s2 += v * v;
    }
    atomicAdd(&stats[c], s);
    atomicAdd(&stats[Hh + c], s2);
}

__global__ void bn_apply_kernel(float* __restrict__ a, const float* __restrict__ stats,
                                const float* __restrict__ gamma, const float* __restrict__ beta,
                                int doRelu) {
    int i = blockIdx.x;       // node
    int c = threadIdx.x;      // channel
    const float inv_n = 1.0f / (float)Nn;
    float mu  = stats[c] * inv_n;
    float var = stats[Hh + c] * inv_n - mu * mu;
    float v = a[(size_t)i * Hh + c];
    if (doRelu) v = fmaxf(v, 0.f);
    a[(size_t)i * Hh + c] = (v - mu) * rsqrtf(var + EPSb) * gamma[c] + beta[c];
}

// ---------------------------------------------------------------------------
// Pool + head
// ---------------------------------------------------------------------------
__global__ void count_kernel(const int* __restrict__ batch, float* __restrict__ cnt) {
    int i = blockIdx.x * 256 + threadIdx.x;
    if (i < Nn) atomicAdd(&cnt[batch[i]], 1.0f);
}

__global__ void pool_kernel(const float* __restrict__ a, const int* __restrict__ batch,
                            float* __restrict__ pooled) {
    int i = blockIdx.x;
    int c = threadIdx.x;
    atomicAdd(&pooled[(size_t)batch[i] * Hh + c], a[(size_t)i * Hh + c]);
}

// z[g,c] = pooled[g,c]/max(cnt,1) + seq[g]·fcW[:,c] + fcb[c]
__global__ void fc_combine_kernel(const float* __restrict__ seq, const float* __restrict__ fcW,
                                  const float* __restrict__ fcb, const float* __restrict__ pooled,
                                  const float* __restrict__ cnt, float* __restrict__ z) {
    int g = blockIdx.x;       // graph
    int c = threadIdx.x;      // channel
    float acc = fcb[c];
    for (int k = 0; k < FIN; ++k)
        acc += seq[(size_t)g * FIN + k] * fcW[(size_t)k * Hh + c];
    float ct = cnt[g];
    ct = ct < 1.f ? 1.f : ct;
    z[(size_t)g * Hh + c] = pooled[(size_t)g * Hh + c] / ct + acc;
}

// out[g,c] = sigmoid(z[g]·linW[:,c] + linb[c])
__global__ void classify_kernel(const float* __restrict__ z, const float* __restrict__ linW,
                                const float* __restrict__ linb, float* __restrict__ out) {
    int idx = blockIdx.x * 256 + threadIdx.x;
    if (idx >= Gg * Cc) return;
    int g = idx / Cc;
    int c = idx - g * Cc;
    float acc = linb[c];
    for (int k = 0; k < Hh; ++k)
        acc += z[(size_t)g * Hh + k] * linW[(size_t)k * Cc + c];
    out[idx] = 1.0f / (1.0f + __expf(-acc));
}

// ---------------------------------------------------------------------------
// Launch
// ---------------------------------------------------------------------------
extern "C" void kernel_launch(void* const* d_in, const int* in_sizes, int n_in,
                              void* d_out, int out_size, void* d_ws, size_t ws_size,
                              hipStream_t stream) {
    const float* x     = (const float*)d_in[0];
    const int*   ei    = (const int*)d_in[1];
    const int*   src   = ei;
    const int*   dst   = ei + Ee;
    const int*   batch = (const int*)d_in[2];
    const float* seq   = (const float*)d_in[3];
    const float* Ws[3] = {(const float*)d_in[4], (const float*)d_in[6], (const float*)d_in[8]};
    const float* bs[3] = {(const float*)d_in[5], (const float*)d_in[7], (const float*)d_in[9]};
    const float* gs[3] = {(const float*)d_in[10], (const float*)d_in[12], (const float*)d_in[14]};
    const float* be[3] = {(const float*)d_in[11], (const float*)d_in[13], (const float*)d_in[15]};
    const float* fcW   = (const float*)d_in[16];
    const float* fcb   = (const float*)d_in[17];
    const float* linW  = (const float*)d_in[18];
    const float* linb  = (const float*)d_in[19];
    float*       out   = (float*)d_out;

    // Workspace layout (floats). buf0+buf1 = 102.4 MB -> both fit in 192 MB L2.
    float* ws     = (float*)d_ws;
    float* buf0   = ws;                                   // N*H  (GEMM out)
    float* buf1   = buf0 + (size_t)Nn * Hh;               // N*H  (agg / layer out)
    float* deg    = buf1 + (size_t)Nn * Hh;               // N    (-> dinv in place)
    float* stats  = deg + Nn;                             // 2*H
    float* pooled = stats + 2 * Hh;                       // G*H
    float* cnt    = pooled + (size_t)Gg * Hh;             // G   (contiguous w/ pooled)
    float* zG     = cnt + Gg;                             // G*H
    _Float16* WhT = (_Float16*)(zG + (size_t)Gg * Hh);    // Hh x K halves (max K=FIN)

    // --- re-zero per call (graph replay safe) ---
    zero_kernel<<<(Nn + 255) / 256, 256, 0, stream>>>(deg, Nn);
    zero_kernel<<<(Gg * Hh + Gg + 255) / 256, 256, 0, stream>>>(pooled, Gg * Hh + Gg);

    // --- degree -> dinv ---
    deg_kernel<<<(Ee + 255) / 256, 256, 0, stream>>>(dst, deg);
    dinv_kernel<<<(Nn + 255) / 256, 256, 0, stream>>>(deg);

    // --- 3 GCN layers ---
    const int Ks[3] = {FIN, Hh, Hh};
    const float* Ain = x;
    for (int l = 0; l < 3; ++l) {
        const int K = Ks[l];
        cvt_f16_T_kernel<<<(Hh * K + 255) / 256, 256, 0, stream>>>(Ws[l], WhT, K);
        dim3 gg((Nn + BM - 1) / BM, Hh / BN);
        gemm_wmma_f16<<<gg, 256, 0, stream>>>(Ain, WhT, buf0, Nn, K, Hh);
        init_agg_kernel<<<Nn, Hh, 0, stream>>>(buf0, deg, bs[l], buf1);
        scatter_kernel<<<(Ee * (Hh / 4) + 255) / 256, 256, 0, stream>>>(src, dst, deg, buf0, buf1);
        zero_kernel<<<(2 * Hh + 255) / 256, 256, 0, stream>>>(stats, 2 * Hh);
        int doRelu = (l < 2) ? 1 : 0;
        bn_stats_kernel<<<(Nn + BN_ROWS - 1) / BN_ROWS, Hh, 0, stream>>>(buf1, stats, doRelu);
        bn_apply_kernel<<<Nn, Hh, 0, stream>>>(buf1, stats, gs[l], be[l], doRelu);
        Ain = buf1;
    }

    // --- pool + head ---
    count_kernel<<<(Nn + 255) / 256, 256, 0, stream>>>(batch, cnt);
    pool_kernel<<<Nn, Hh, 0, stream>>>(buf1, batch, pooled);
    fc_combine_kernel<<<Gg, Hh, 0, stream>>>(seq, fcW, fcb, pooled, cnt, zG);
    classify_kernel<<<(Gg * Cc + 255) / 256, 256, 0, stream>>>(zG, linW, linb, out);
}
